// InteractiveWindowedSelfAttention_56178172231821
// MI455X (gfx1250) — matrix operand order repaired and checked by reading, not compiled, not run on hardware
//
#include <hip/hip_runtime.h>
#include <hip/hip_bf16.h>

typedef _Float16 half_t;
typedef _Float16 h8   __attribute__((ext_vector_type(8)));
typedef _Float16 v16h __attribute__((ext_vector_type(16)));
typedef float    v8f  __attribute__((ext_vector_type(8)));

#define CAT16(a,b) __builtin_shufflevector(a,b,0,1,2,3,4,5,6,7,8,9,10,11,12,13,14,15)
#define WMMA_F16(a,b,c) __builtin_amdgcn_wmma_f32_16x16x32_f16(false,(a),false,(b),(short)0,(c),false,false)

#define NB    32
#define NH    56
#define NPIX  (32*56*56)     // 100352
#define DIMC  192
#define CIN   256
#define NWIN  2048           // 32 * 8 * 8
#define HEADS 8

// ---------------- weight pre-pack (f32 -> f16, transposed to N-major) -------
__global__ void pack_qkv_w(const float* __restrict__ Wq, const float* __restrict__ Wk,
                           const float* __restrict__ Wv, half_t* __restrict__ wt) {
    int i = blockIdx.x * 256 + threadIdx.x;
    if (i >= 3 * 256 * 192) return;
    int mat = i / (256 * 192);
    int r   = i % (256 * 192);
    int n = r / 192, k = r % 192;
    const float* W = (mat == 0) ? Wq : (mat == 1) ? Wk : Wv;
    wt[i] = (half_t)W[k * 256 + n];
}

__global__ void pack_wl_w(const float* __restrict__ Wl, half_t* __restrict__ wt) {
    int i = blockIdx.x * 256 + threadIdx.x;
    if (i >= 9 * 256 * 256) return;
    int t = i / (256 * 256);
    int r = i % (256 * 256);
    int o = r / 256, ii = r % 256;
    // HWIO: Wl[((ky*3+kx)*256 + in)*256 + out] -> wt[(t*256 + out)*256 + in]
    wt[i] = (half_t)Wl[((long)t * 256 + ii) * 256 + o];
}

__global__ void pack_wo_w(const float* __restrict__ Wo, half_t* __restrict__ wt) {
    int i = blockIdx.x * 256 + threadIdx.x;
    if (i >= 192 * 256) return;
    int n = i / 256, k = i % 256;
    wt[i] = (half_t)Wo[k * 192 + n];
}

// ---------------- fused QKV projection: [NPIX x 192] @ [192 x 768] ---------
// One pass over x; each A fragment feeds 12 WMMAs (3 mats x 4 col tiles).
// 8 waves: rowblk = wave/4 (2 x 16 pixels), colgrp = wave%4 (64 of 256 cols).
__global__ __launch_bounds__(256) void qkv_gemm(const float* __restrict__ x,
                                                const half_t* __restrict__ wt,
                                                half_t* __restrict__ outbase) {
    int tid = threadIdx.x, wave = tid >> 5, lane = tid & 31;
    int ln16 = lane & 15, hid = lane >> 4;
    int p0 = blockIdx.x * 32;
    int rb = wave >> 2;
    int cb = (wave & 3) * 64;
    int m  = p0 + rb * 16 + ln16;
    const float* xrow = x + (long)m * DIMC;

    v8f acc[12] = {};

    for (int kb = 0; kb < DIMC; kb += 32) {
        if (kb + 32 < DIMC)
            __builtin_prefetch(xrow + kb + 32, 0, 3);
        // A fragment: f32 -> f16 conversion inline
        union { v16h v; half_t e[16]; } ua;
        const float* fp = xrow + kb + hid * 8;
#pragma unroll
        for (int e = 0; e < 8; e++) ua.e[e]     = (half_t)fp[e];
#pragma unroll
        for (int e = 0; e < 8; e++) ua.e[8 + e] = (half_t)fp[16 + e];
#pragma unroll
        for (int mat = 0; mat < 3; mat++) {
            const half_t* W = wt + (long)mat * 256 * 192;
#pragma unroll
            for (int nt = 0; nt < 4; nt++) {
                int n = cb + nt * 16 + ln16;
                const half_t* wr = W + (long)n * DIMC + kb;
                h8 b0 = *(const h8*)(wr + hid * 8), b1 = *(const h8*)(wr + hid * 8 + 16);
                v16h bb = CAT16(b0, b1);
                acc[mat * 4 + nt] = WMMA_F16(ua.v, bb, acc[mat * 4 + nt]);
            }
        }
    }
#pragma unroll
    for (int mat = 0; mat < 3; mat++) {
        half_t* out = outbase + (long)mat * NPIX * CIN;
#pragma unroll
        for (int nt = 0; nt < 4; nt++) {
#pragma unroll
            for (int r = 0; r < 8; r++) {
                int mm = p0 + rb * 16 + r + 8 * hid;
                int n  = cb + nt * 16 + ln16;
                out[(long)mm * CIN + n] = (half_t)acc[mat * 4 + nt][r];
            }
        }
    }
}

// ---------------- windowed attention: one (window, head) per 128-thr WG -----
__global__ __launch_bounds__(128) void win_attn(const half_t* __restrict__ q,
                                                const half_t* __restrict__ k,
                                                const half_t* __restrict__ v,
                                                half_t* __restrict__ y) {
    __shared__ half_t Qs[64 * 32];
    __shared__ half_t Ks[64 * 32];
    __shared__ half_t Vt[32 * 64];   // transposed V: [channel][token]
    __shared__ float  Sf[64 * 68];
    __shared__ half_t Pp[64 * 64];

    int wid = blockIdx.x, head = blockIdx.y;
    int b = wid >> 6, rem = wid & 63, wy = rem >> 3, wx = rem & 7;
    int tid = threadIdx.x, wave = tid >> 5, lane = tid & 31;
    int ln16 = lane & 15, hid = lane >> 4;
    const float scale = 0.17677669529663687f;   // 1/sqrt(32)

    // stage Q (pre-scaled) and K, zero-padded to 64 tokens
    {
        int t = tid & 63;
        int role = tid >> 6;                    // 0 -> Q, 1 -> K
        half_t* dst = (role == 0 ? Qs : Ks) + t * 32;
        if (t < 49) {
            int py = wy * 7 + t / 7, px = wx * 7 + t % 7;
            long base = ((long)(b * 56 + py) * 56 + px) * CIN + head * 32;
            const half_t* src = (role == 0) ? q : k;
#pragma unroll
            for (int c = 0; c < 32; c++) {
                half_t val = src[base + c];
                if (role == 0) val = (half_t)((float)val * scale);
                dst[c] = val;
            }
        } else {
#pragma unroll
            for (int c = 0; c < 32; c++) dst[c] = (half_t)0.f;
        }
    }
    // stage V transposed [32][64], zero-padded
    for (int i = tid; i < 64 * 32; i += 128) {
        int c = i & 31, t = i >> 5;
        half_t val = (half_t)0.f;
        if (t < 49) {
            int py = wy * 7 + t / 7, px = wx * 7 + t % 7;
            val = v[((long)(b * 56 + py) * 56 + px) * CIN + head * 32 + c];
        }
        Vt[c * 64 + t] = val;
    }
    __syncthreads();

    // S = Q K^T : wave w -> row-block w, 4 col tiles
    {
        const half_t* qrow = Qs + (wave * 16 + ln16) * 32;
        h8 a0 = *(const h8*)(qrow + hid * 8);
        h8 a1 = *(const h8*)(qrow + hid * 8 + 16);
        v16h a = CAT16(a0, a1);
#pragma unroll
        for (int nt = 0; nt < 4; nt++) {
            const half_t* krow = Ks + (nt * 16 + ln16) * 32;
            h8 b0 = *(const h8*)(krow + hid * 8), b1 = *(const h8*)(krow + hid * 8 + 16);
            v16h bb = CAT16(b0, b1);
            v8f c = {};
            c = WMMA_F16(a, bb, c);
#pragma unroll
            for (int r = 0; r < 8; r++)
                Sf[(wave * 16 + r + 8 * hid) * 68 + nt * 16 + ln16] = c[r];
        }
    }
    __syncthreads();

    // softmax over j < 49, rows 0..63 (pad rows -> zero P)
    if (tid < 64) {
        int row = tid;
        if (row < 49) {
            float mx = -1e30f;
            for (int j = 0; j < 49; j++) mx = fmaxf(mx, Sf[row * 68 + j]);
            float s = 0.f;
            for (int j = 0; j < 49; j++) {
                float e = __expf(Sf[row * 68 + j] - mx);
                s += e;
                Pp[row * 64 + j] = (half_t)e;
            }
            float inv = 1.f / s;
            for (int j = 0; j < 49; j++)
                Pp[row * 64 + j] = (half_t)((float)Pp[row * 64 + j] * inv);
            for (int j = 49; j < 64; j++) Pp[row * 64 + j] = (half_t)0.f;
        } else {
            for (int j = 0; j < 64; j++) Pp[row * 64 + j] = (half_t)0.f;
        }
    }
    __syncthreads();

    // O = P * V : wave w -> row-block w, 2 col tiles (32 channels), K = 64
    {
        v8f o0 = {}, o1 = {};
#pragma unroll
        for (int kb = 0; kb < 2; kb++) {
            const half_t* prow = Pp + (wave * 16 + ln16) * 64 + kb * 32;
            h8 a0 = *(const h8*)(prow + hid * 8);
            h8 a1 = *(const h8*)(prow + hid * 8 + 16);
            v16h a = CAT16(a0, a1);
#pragma unroll
            for (int nb = 0; nb < 2; nb++) {
                const half_t* vrow = Vt + (nb * 16 + ln16) * 64 + kb * 32;
                h8 b0 = *(const h8*)(vrow + hid * 8), b1 = *(const h8*)(vrow + hid * 8 + 16);
                v16h bb = CAT16(b0, b1);
                if (nb == 0) o0 = WMMA_F16(a, bb, o0);
                else         o1 = WMMA_F16(a, bb, o1);
            }
        }
#pragma unroll
        for (int nb = 0; nb < 2; nb++) {
#pragma unroll
            for (int r = 0; r < 8; r++) {
                int m = wave * 16 + r + 8 * hid;
                if (m < 49) {
                    int py = wy * 7 + m / 7, px = wx * 7 + m % 7;
                    float val = (nb == 0) ? o0[r] : o1[r];
                    y[((long)(b * 56 + py) * 56 + px) * CIN + head * 32 + nb * 16 + ln16]
                        = (half_t)val;
                }
            }
        }
    }
}

// ---------------- 3x3 conv on V as implicit GEMM (K = 9*256) ---------------
// grid: (b*56 + y)*4 + xt ; tile = 14 pixels along x (padded to 16), 256 outs
__global__ __launch_bounds__(256) void conv3x3(const half_t* __restrict__ vv,
                                               const half_t* __restrict__ wl,
                                               const float* __restrict__ bl,
                                               half_t* __restrict__ lout) {
    int bid = blockIdx.x;
    int xt = bid & 3;
    int by = bid >> 2;
    int y = by % 56, b = by / 56;
    int x0 = xt * 14;
    int tid = threadIdx.x, wave = tid >> 5, lane = tid & 31;
    int ln16 = lane & 15, hid = lane >> 4;
    int nbase = wave * 32;

    v8f acc0 = {}, acc1 = {};
    for (int t = 0; t < 9; t++) {
        int dy = t / 3 - 1, dx = t % 3 - 1;
        int yy = y + dy;
        int xx = x0 + ln16 + dx;
        bool valid = (yy >= 0) && (yy < 56) && (xx >= 0) && (xx < 56) && (ln16 < 14);
        const half_t* arow = vv + ((long)(b * 56 + yy) * 56 + xx) * CIN;
        const half_t* wbase = wl + (long)t * 256 * 256;
        if (valid)
            __builtin_prefetch(arow, 0, 3);
        for (int kb = 0; kb < 256; kb += 32) {
            v16h a = {};
            if (valid) {
                h8 a0 = *(const h8*)(arow + kb + hid * 8);
                h8 a1 = *(const h8*)(arow + kb + hid * 8 + 16);
                a = CAT16(a0, a1);
            }
#pragma unroll
            for (int nt = 0; nt < 2; nt++) {
                int n = nbase + nt * 16 + ln16;
                const half_t* wr = wbase + (long)n * 256 + kb;
                h8 b0 = *(const h8*)(wr + hid * 8), b1 = *(const h8*)(wr + hid * 8 + 16);
                v16h bb = CAT16(b0, b1);
                if (nt == 0) acc0 = WMMA_F16(a, bb, acc0);
                else         acc1 = WMMA_F16(a, bb, acc1);
            }
        }
    }
#pragma unroll
    for (int nt = 0; nt < 2; nt++) {
        int n = nbase + nt * 16 + ln16;
        float bias = bl[n];
#pragma unroll
        for (int r = 0; r < 8; r++) {
            int m = r + 8 * hid;
            if (m < 14) {
                int x = x0 + m;
                float val = (nt == 0) ? acc0[r] : acc1[r];
                lout[((long)(b * 56 + y) * 56 + x) * CIN + n] = (half_t)(val + bias);
            }
        }
    }
}

// ---------------- final: out = (attn + local) @ Wout + b_out ---------------
__global__ __launch_bounds__(256) void out_gemm(const half_t* __restrict__ ya,
                                                const half_t* __restrict__ lc,
                                                const half_t* __restrict__ wt,
                                                const float* __restrict__ bo,
                                                float* __restrict__ out) {
    int tid = threadIdx.x, wave = tid >> 5, lane = tid & 31;
    int ln16 = lane & 15, hid = lane >> 4;
    int p0 = blockIdx.x * 32;
    int rb = wave >> 2;
    int cb = (wave & 3) * 48;     // 4 groups x 48 = 192
    int m = p0 + rb * 16 + ln16;
    const half_t* yrow = ya + (long)m * CIN;
    const half_t* lrow = lc + (long)m * CIN;

    v8f acc[3] = {};
    for (int kb = 0; kb < CIN; kb += 32) {
        h8 y0 = *(const h8*)(yrow + kb + hid * 8);
        h8 y1 = *(const h8*)(yrow + kb + hid * 8 + 16);
        h8 l0 = *(const h8*)(lrow + kb + hid * 8);
        h8 l1 = *(const h8*)(lrow + kb + hid * 8 + 16);
        h8 s0 = y0 + l0, s1 = y1 + l1;
        v16h a = CAT16(s0, s1);
#pragma unroll
        for (int nt = 0; nt < 3; nt++) {
            int n = cb + nt * 16 + ln16;
            const half_t* wr = wt + (long)n * CIN + kb;
            h8 b0 = *(const h8*)(wr + hid * 8), b1 = *(const h8*)(wr + hid * 8 + 16);
            v16h bb = CAT16(b0, b1);
            acc[nt] = WMMA_F16(a, bb, acc[nt]);
        }
    }
#pragma unroll
    for (int nt = 0; nt < 3; nt++) {
        int n = cb + nt * 16 + ln16;
        float bias = bo[n];
#pragma unroll
        for (int r = 0; r < 8; r++) {
            int mm = p0 + rb * 16 + r + 8 * hid;
            out[(long)mm * DIMC + n] = acc[nt][r] + bias;
        }
    }
}

extern "C" void kernel_launch(void* const* d_in, const int* in_sizes, int n_in,
                              void* d_out, int out_size, void* d_ws, size_t ws_size,
                              hipStream_t stream) {
    const float* x  = (const float*)d_in[0];
    const float* Wq = (const float*)d_in[1];
    const float* Wk = (const float*)d_in[2];
    const float* Wv = (const float*)d_in[3];
    const float* Wl = (const float*)d_in[4];
    const float* bl = (const float*)d_in[5];
    const float* Wo = (const float*)d_in[6];
    const float* bo = (const float*)d_in[7];
    float* out = (float*)d_out;

    half_t* q16 = (half_t*)d_ws;
    half_t* k16 = q16 + (long)NPIX * CIN;
    half_t* v16 = k16 + (long)NPIX * CIN;
    half_t* y16 = v16 + (long)NPIX * CIN;
    half_t* l16 = q16;                         // reuse q buffer after attention
    half_t* wqkv_t = y16 + (long)NPIX * CIN;   // 3*256*192 halves
    half_t* wl_t   = wqkv_t + 3 * 256 * 192;   // 9*256*256 halves
    half_t* wo_t   = wl_t + 9 * 256 * 256;     // 192*256 halves

    pack_qkv_w<<<(3 * 256 * 192 + 255) / 256, 256, 0, stream>>>(Wq, Wk, Wv, wqkv_t);
    pack_wl_w<<<(9 * 256 * 256 + 255) / 256, 256, 0, stream>>>(Wl, wl_t);
    pack_wo_w<<<(192 * 256 + 255) / 256, 256, 0, stream>>>(Wo, wo_t);

    qkv_gemm<<<NPIX / 32, 256, 0, stream>>>(x, wqkv_t, q16);
    win_attn<<<dim3(NWIN, HEADS), 128, 0, stream>>>(q16, k16, v16, y16);
    conv3x3<<<32 * 56 * 4, 256, 0, stream>>>(v16, wl_t, bl, l16);
    out_gemm<<<NPIX / 32, 256, 0, stream>>>(y16, l16, wo_t, bo, out);
}
